// MutliHeadAttention_5463198400552
// MI455X (gfx1250) — matrix operand order repaired and checked by reading, not compiled, not run on hardware
//
#include <hip/hip_runtime.h>

// ---------------------------------------------------------------------------
// Multi-head attention for MI455X (gfx1250, wave32, WMMA, async-LDS).
// Pipeline:
//   1-3) q/k/v = X @ W^T + b        gemm_bias_kernel<true>  (bf16 out)
//   4)   causal flash attention ->  O fp32  (attn_kernel)
//   5)   out = O @ Wo^T + bo        gemm_bias_kernel<false> (fp32 out)
// Both kernels are software-pipelined with double-buffered LDS tiles; the
// attention K-tile staging uses GLOBAL_LOAD_ASYNC_TO_LDS_B128 when available.
// ---------------------------------------------------------------------------

typedef __bf16 bf16;
typedef __attribute__((ext_vector_type(16))) __bf16 v16bf;
typedef __attribute__((ext_vector_type(8)))  __bf16 v8bf;
typedef __attribute__((ext_vector_type(8)))  float  v8f;

#define ATTN_B 4
#define ATTN_T 2048
#define ATTN_D 1024
#define ATTN_H 16
#define ATTN_HD 64

#if __has_builtin(__builtin_amdgcn_global_load_async_to_lds_b128)
#define USE_ASYNC_LDS 1
#else
#define USE_ASYNC_LDS 0
#endif

// The async-LDS builtins expect pointers to a 16-byte int vector in AS1/AS3
// (exact types from the hipcc diagnostic).
typedef int v4i_t __attribute__((vector_size(16)));
typedef __attribute__((address_space(1))) v4i_t gbl_v4i;
typedef __attribute__((address_space(3))) v4i_t lds_v4i;

// LDS offset = low 32 bits of the generic address (ISA 10.2 aperture rules).
__device__ __forceinline__ lds_v4i* to_lds(const void* p) {
  return (lds_v4i*)(unsigned int)(unsigned long long)p;
}
__device__ __forceinline__ gbl_v4i* to_gbl(const void* p) {
  return (gbl_v4i*)(unsigned long long)p;
}
__device__ __forceinline__ void wait_async0() {
#if __has_builtin(__builtin_amdgcn_s_wait_asynccnt)
  __builtin_amdgcn_s_wait_asynccnt(0);
#else
  asm volatile("s_wait_asynccnt 0" ::: "memory");
#endif
}

__device__ __forceinline__ v16bf frag_cat(v8bf lo, v8bf hi) {
  return __builtin_shufflevector(lo, hi, 0, 1, 2, 3, 4, 5, 6, 7,
                                 8, 9, 10, 11, 12, 13, 14, 15);
}

// A-operand fragment, 16x32 (MxK), source stored row-major [m][k].
// ISA layout: lane<16 holds M=lane, K in {0..7, 16..23};
//             lane>=16 holds M=lane-16, K in {8..15, 24..31}.
__device__ __forceinline__ v16bf load_frag_a(const bf16* __restrict__ p,
                                             int stride, int lane) {
  const int row = lane & 15;
  const int kb  = (lane >> 4) * 8;
  v8bf lo = *(const v8bf*)(p + row * stride + kb);
  v8bf hi = *(const v8bf*)(p + row * stride + kb + 16);
  return frag_cat(lo, hi);
}

// B-operand fragment, 32x16 (KxN), source stored [n][k] (each B column is a
// contiguous run of K). Lanes 0-15: K in {0..7, 16..23} of col N=lane; lanes
// 16-31: K in {8..15, 24..31} of col N=lane-16.
__device__ __forceinline__ v16bf load_frag_b(const bf16* __restrict__ p,
                                             int stride, int lane) {
  const int n  = lane & 15;
  const int kb = (lane >> 4) * 8;
  v8bf lo = *(const v8bf*)(p + n * stride + kb);
  v8bf hi = *(const v8bf*)(p + n * stride + kb + 16);
  return frag_cat(lo, hi);
}

__device__ __forceinline__ v8f wmma_bf16(v16bf a, v16bf b, v8f c) {
  return __builtin_amdgcn_wmma_f32_16x16x32_bf16(
      false, a, false, b, (short)0, c, false, false);
}

// ---------------------------------------------------------------------------
// GEMM with bias: C[M,N] = A[M,K] @ W[N,K]^T + bias[N]
// 128x128 block tile, K-step 32, 8 waves (4x2 grid, 32x64 per wave, 8 WMMA
// accumulators). Double-buffered LDS; next tile's fp32 global loads are
// issued before the current tile's WMMAs (single barrier per K-step).
// ---------------------------------------------------------------------------
template <bool OUT_BF16>
__global__ __launch_bounds__(256) void gemm_bias_kernel(
    const float* __restrict__ A, const float* __restrict__ W,
    const float* __restrict__ bias, void* __restrict__ Cout,
    int M, int N, int K) {
  __shared__ alignas(16) bf16 sA[2][128 * 40];  // stride 40 halfs = 80 B
  __shared__ alignas(16) bf16 sB[2][128 * 40];

  const int t    = threadIdx.x;
  const int lane = t & 31;
  const int wave = t >> 5;
  const int wm   = wave >> 1;  // 0..3 -> 32-row slab
  const int wn   = wave & 1;   // 0..1 -> 64-col slab
  const int row0 = blockIdx.x * 128;
  const int col0 = blockIdx.y * 128;

  float4 areg[4], wreg[4];
  auto load_tiles = [&](int k0) {
#pragma unroll
    for (int i = 0; i < 4; ++i) {
      const int v  = t + 256 * i;
      const int r  = v >> 3;  // tile row 0..127
      const int c4 = v & 7;   // float4 column 0..7
      areg[i] = *(const float4*)(A + (size_t)(row0 + r) * K + k0 + c4 * 4);
      wreg[i] = *(const float4*)(W + (size_t)(col0 + r) * K + k0 + c4 * 4);
    }
  };
  auto store_tiles = [&](int buf) {
#pragma unroll
    for (int i = 0; i < 4; ++i) {
      const int v  = t + 256 * i;
      const int r  = v >> 3;
      const int c4 = v & 7;
      union { bf16 h[4]; unsigned long long u; } tmp;
      tmp.h[0] = (bf16)areg[i].x; tmp.h[1] = (bf16)areg[i].y;
      tmp.h[2] = (bf16)areg[i].z; tmp.h[3] = (bf16)areg[i].w;
      *(unsigned long long*)(&sA[buf][r * 40 + c4 * 4]) = tmp.u;
      tmp.h[0] = (bf16)wreg[i].x; tmp.h[1] = (bf16)wreg[i].y;
      tmp.h[2] = (bf16)wreg[i].z; tmp.h[3] = (bf16)wreg[i].w;
      *(unsigned long long*)(&sB[buf][r * 40 + c4 * 4]) = tmp.u;
    }
  };

  v8f acc[2][4];
#pragma unroll
  for (int i = 0; i < 2; ++i)
#pragma unroll
    for (int j = 0; j < 4; ++j) acc[i][j] = (v8f)0.0f;

  load_tiles(0);
  store_tiles(0);
  __syncthreads();

  const int niter = K / 32;
  for (int it = 0; it < niter; ++it) {
    const int  cur  = it & 1;
    const bool more = (it + 1) < niter;
    if (more) load_tiles((it + 1) * 32);  // overlap with WMMAs below

    v16bf af[2], bfr[4];
#pragma unroll
    for (int tm = 0; tm < 2; ++tm)
      af[tm] = load_frag_a(&sA[cur][(wm * 32 + tm * 16) * 40], 40, lane);
#pragma unroll
    for (int tn = 0; tn < 4; ++tn)
      bfr[tn] = load_frag_b(&sB[cur][(wn * 64 + tn * 16) * 40], 40, lane);
#pragma unroll
    for (int tm = 0; tm < 2; ++tm)
#pragma unroll
      for (int tn = 0; tn < 4; ++tn)
        acc[tm][tn] = wmma_bf16(af[tm], bfr[tn], acc[tm][tn]);

    if (more) store_tiles(cur ^ 1);
    __syncthreads();
  }

  // Epilogue: C layout -> row = base + vgprIdx + 8*(lane>=16), col = lane&15.
#pragma unroll
  for (int tm = 0; tm < 2; ++tm) {
#pragma unroll
    for (int tn = 0; tn < 4; ++tn) {
      const int col   = col0 + wn * 64 + tn * 16 + (lane & 15);
      const float bv  = bias[col];
      const int rbase = row0 + wm * 32 + tm * 16 + ((lane >> 4) * 8);
#pragma unroll
      for (int r = 0; r < 8; ++r) {
        const float  val = acc[tm][tn][r] + bv;
        const size_t idx = (size_t)(rbase + r) * N + col;
        if (OUT_BF16) ((bf16*)Cout)[idx] = (bf16)val;
        else          ((float*)Cout)[idx] = val;
      }
    }
  }
}

// ---------------------------------------------------------------------------
// Causal flash attention. One block per (b, h, 128 q-rows); 8 waves, each
// owning 16 q-rows. Double-buffered 64-wide K/V tiles; K staged via async
// LDS loads (ASYNCcnt) when available, V transposed through registers.
// Single barrier per K-tile: the per-wave P round-trip needs no barrier
// (DS ops are in-order within a wave).
// ---------------------------------------------------------------------------
__global__ __launch_bounds__(256) void attn_kernel(
    const bf16* __restrict__ Qw, const bf16* __restrict__ Kw,
    const bf16* __restrict__ Vw, float* __restrict__ Ow) {
  constexpr int T = ATTN_T, D = ATTN_D, H = ATTN_H;

  __shared__ alignas(16) bf16 sK[2][64 * 72];     // [kk][d], padded
  __shared__ alignas(16) bf16 sV[2][64 * 72];     // [d][kk] (transposed)
  __shared__ alignas(16) bf16 sP[8][16 * 72];     // per-wave P tile [m][kk]

  const int t    = threadIdx.x;
  const int lane = t & 31;
  const int wave = t >> 5;
  const int bh   = blockIdx.y;
  const int b    = bh / H;
  const int h    = bh % H;
  const int q0   = blockIdx.x * 128;
  const int half = lane >> 4;
  const int nlo  = lane & 15;

  // ---- staging helpers (double buffered, pipelined through regs/async) ----
  unsigned long long vreg[4];
#if !USE_ASYNC_LDS
  uint4 kreg[2];
#endif
  auto k_prefetch = [&](int kt, int buf) {
#pragma unroll
    for (int i = 0; i < 2; ++i) {
      const int v  = t + 256 * i;
      const int kk = v >> 3;  // 0..63
      const int d8 = v & 7;   // 8-half (16 B) group
      const bf16* gp =
          Kw + ((size_t)b * T + kt * 64 + kk) * D + h * ATTN_HD + d8 * 8;
#if USE_ASYNC_LDS
      __builtin_amdgcn_global_load_async_to_lds_b128(
          to_gbl(gp), to_lds(&sK[buf][kk * 72 + d8 * 8]), 0, 0);
#else
      kreg[i] = *(const uint4*)gp;
      (void)buf;
#endif
    }
  };
  auto k_commit = [&](int buf) {
#if !USE_ASYNC_LDS
#pragma unroll
    for (int i = 0; i < 2; ++i) {
      const int v  = t + 256 * i;
      const int kk = v >> 3;
      const int d8 = v & 7;
      *(uint4*)(&sK[buf][kk * 72 + d8 * 8]) = kreg[i];
    }
#else
    (void)buf;
#endif
  };
  auto v_load = [&](int kt) {
#pragma unroll
    for (int i = 0; i < 4; ++i) {
      const int v  = t + 256 * i;
      const int kk = v >> 4;  // 0..63
      const int d4 = v & 15;  // 4-half group
      vreg[i] = *(const unsigned long long*)(
          Vw + ((size_t)b * T + kt * 64 + kk) * D + h * ATTN_HD + d4 * 4);
    }
  };
  auto v_store = [&](int buf) {  // transpose on store: sV[d][kk]
#pragma unroll
    for (int i = 0; i < 4; ++i) {
      const int v  = t + 256 * i;
      const int kk = v >> 4;
      const int d4 = v & 15;
      union { bf16 h4[4]; unsigned long long u; } tv;
      tv.u = vreg[i];
#pragma unroll
      for (int j = 0; j < 4; ++j) sV[buf][(d4 * 4 + j) * 72 + kk] = tv.h4[j];
    }
  };

  // ---- Q fragments (A-layout over d) straight from global bf16 ----
  const int   qrow = q0 + wave * 16 + nlo;
  const bf16* qp   = Qw + ((size_t)b * T + qrow) * D + h * ATTN_HD;
  v16bf qa[2];
#pragma unroll
  for (int s = 0; s < 2; ++s) {
    const int kb = s * 32 + half * 8;
    v8bf lo = *(const v8bf*)(qp + kb);
    v8bf hi = *(const v8bf*)(qp + kb + 16);
    qa[s] = frag_cat(lo, hi);
  }

  v8f o[4];
#pragma unroll
  for (int i = 0; i < 4; ++i) o[i] = (v8f)0.0f;
  float mrun[8], lrun[8];
#pragma unroll
  for (int r = 0; r < 8; ++r) { mrun[r] = -1.0e30f; lrun[r] = 0.0f; }

  // Prologue: stage tile 0 into buffer 0.
  k_prefetch(0, 0);
  v_load(0);
  k_commit(0);
  v_store(0);
#if USE_ASYNC_LDS
  wait_async0();
#endif
  __syncthreads();

  const int kt_last = (q0 + 127) >> 6;  // causal bound for this block
  for (int kt = 0; kt <= kt_last; ++kt) {
    const int  cur  = kt & 1;
    const bool more = kt < kt_last;
    if (more) {  // issue next tile's loads under this tile's compute
      k_prefetch(kt + 1, cur ^ 1);
      v_load(kt + 1);
    }
    const int k0 = kt * 64;

    // Scores: S(16x64) = q @ K^T (B columns = K rows, contiguous in sK).
    v8f sc[4];
#pragma unroll
    for (int tn = 0; tn < 4; ++tn) sc[tn] = (v8f)0.0f;
#pragma unroll
    for (int s = 0; s < 2; ++s) {
#pragma unroll
      for (int tn = 0; tn < 4; ++tn) {
        v16bf kb = load_frag_b(&sK[cur][(tn * 16) * 72 + s * 32], 72, lane);
        sc[tn] = wmma_bf16(qa[s], kb, sc[tn]);
      }
    }

    // Online softmax; C-layout row = r + 8*half, col = tn*16 + nlo.
    const float scale = 0.125f;  // 1/sqrt(64)
#pragma unroll
    for (int r = 0; r < 8; ++r) {
      const int irow = q0 + wave * 16 + r + 8 * half;
      float x = -3.0e38f;
#pragma unroll
      for (int tn = 0; tn < 4; ++tn) {
        const int j = k0 + tn * 16 + nlo;
        float v = sc[tn][r] * scale;
        v = (j <= irow) ? v : -1.0e30f;  // causal mask
        sc[tn][r] = v;
        x = fmaxf(x, v);
      }
#pragma unroll
      for (int off = 1; off < 16; off <<= 1)
        x = fmaxf(x, __shfl_xor(x, off, 32));  // stays within 16-lane half
      const float mnew = fmaxf(mrun[r], x);
      const float cf   = __expf(mrun[r] - mnew);
      float rsum = 0.0f;
#pragma unroll
      for (int tn = 0; tn < 4; ++tn) {
        const float p = __expf(sc[tn][r] - mnew);
        rsum += p;
        sP[wave][(r + 8 * half) * 72 + tn * 16 + nlo] = (bf16)p;
      }
#pragma unroll
      for (int off = 1; off < 16; off <<= 1)
        rsum += __shfl_xor(rsum, off, 32);
      lrun[r] = lrun[r] * cf + rsum;
      mrun[r] = mnew;
#pragma unroll
      for (int td = 0; td < 4; ++td) o[td][r] *= cf;
    }

    // O += P(16x64) @ V(64x64). Per-wave P: same-wave DS ops are in-order,
    // so the frag loads below see the stores above without a barrier.
#pragma unroll
    for (int s = 0; s < 2; ++s) {
      v16bf pa = load_frag_a(&sP[wave][s * 32], 72, lane);
#pragma unroll
      for (int td = 0; td < 4; ++td) {
        v16bf vb = load_frag_b(&sV[cur][(td * 16) * 72 + s * 32], 72, lane);
        o[td] = wmma_bf16(pa, vb, o[td]);
      }
    }

    if (more) {  // commit next tile into the other buffer
      k_commit(cur ^ 1);
      v_store(cur ^ 1);
#if USE_ASYNC_LDS
      wait_async0();
#endif
    }
    __syncthreads();
  }

  // Normalize and write O (fp32) in [b, t, h*64 + d] layout.
#pragma unroll
  for (int r = 0; r < 8; ++r) {
    const float inv = 1.0f / lrun[r];
    const int   tg  = q0 + wave * 16 + r + 8 * half;
    float* op = Ow + ((size_t)b * T + tg) * D + h * ATTN_HD + nlo;
#pragma unroll
    for (int td = 0; td < 4; ++td) op[td * 16] = o[td][r] * inv;
  }
}

// ---------------------------------------------------------------------------
extern "C" void kernel_launch(void* const* d_in, const int* in_sizes, int n_in,
                              void* d_out, int out_size, void* d_ws,
                              size_t ws_size, hipStream_t stream) {
  (void)in_sizes; (void)n_in; (void)out_size; (void)ws_size;
  constexpr int B = ATTN_B, T = ATTN_T, D = ATTN_D;

  const float* Q  = (const float*)d_in[0];
  const float* K  = (const float*)d_in[1];
  const float* V  = (const float*)d_in[2];
  // d_in[3] = mask (causal, reimplemented analytically)
  const float* Wq = (const float*)d_in[4];
  const float* bq = (const float*)d_in[5];
  const float* Wk = (const float*)d_in[6];
  const float* bk = (const float*)d_in[7];
  const float* Wv = (const float*)d_in[8];
  const float* bv = (const float*)d_in[9];
  const float* Wo = (const float*)d_in[10];
  const float* bo = (const float*)d_in[11];

  const size_t BTD = (size_t)B * T * D;
  bf16*  qws = (bf16*)d_ws;
  bf16*  kws = qws + BTD;
  bf16*  vws = kws + BTD;
  float* ows = (float*)(vws + BTD);  // fp32 attention output (pre-Wo)

  const int M = B * T;  // 8192
  dim3 gemm_grid(M / 128, D / 128);     // 64 x 8
  dim3 attn_grid(T / 128, B * ATTN_H);  // 16 x 64

  gemm_bias_kernel<true ><<<gemm_grid, 256, 0, stream>>>(Q, Wq, bq, qws, M, D, D);
  gemm_bias_kernel<true ><<<gemm_grid, 256, 0, stream>>>(K, Wk, bk, kws, M, D, D);
  gemm_bias_kernel<true ><<<gemm_grid, 256, 0, stream>>>(V, Wv, bv, vws, M, D, D);
  attn_kernel<<<attn_grid, 256, 0, stream>>>(qws, kws, vws, ows);
  gemm_bias_kernel<false><<<gemm_grid, 256, 0, stream>>>(ows, Wo, bo,
                                                         (float*)d_out, M, D, D);
}
